// TransformerEncoder_64106681860560
// MI455X (gfx1250) — matrix operand order repaired and checked by reading, not compiled, not run on hardware
//
#include <hip/hip_runtime.h>
#include <hip/hip_bf16.h>
#include <cstdint>
#include <math.h>

// ---------------------------------------------------------------------------
// TransformerEncoder on MI455X (gfx1250, wave32, WMMA).
// GEMMs: bf16 operands, f32 accumulate via v_wmma_f32_16x16x32_bf16.
// LDS staging via GLOBAL_LOAD_ASYNC_TO_LDS_B128 (ASYNCcnt) + ping-pong buffers.
// All B operands pre-transposed to [N][K] so staging is pure 16B async copies.
// ---------------------------------------------------------------------------

#define N_TOK 4096
#define DMODEL 1024
#define QKV_N 3072   // H*(2*KD+VD) = 16*192
#define MHK_  1024   // H*KD
#define HID_  512

typedef __attribute__((ext_vector_type(16))) __bf16 v16bf;
typedef __attribute__((ext_vector_type(8)))  float  v8f;

__device__ __forceinline__ uint16_t f2bf(float f) {
  uint32_t u = __float_as_uint(f);
  u += 0x7FFFu + ((u >> 16) & 1u);      // round-to-nearest-even
  return (uint16_t)(u >> 16);
}
__device__ __forceinline__ float to_f32(float v) { return v; }
__device__ __forceinline__ float to_f32(uint16_t v) {
  return __uint_as_float((uint32_t)v << 16);
}
__device__ __forceinline__ float gelu_exact(float x) {
  return 0.5f * x * (1.0f + erff(x * 0.70710678118654752f));
}

// ---- CDNA5 async global->LDS copy (16B per lane), tracked by ASYNCcnt ----
__device__ __forceinline__ void async_copy_b128(uint32_t lds_byte_off,
                                                const void* gaddr) {
  asm volatile("global_load_async_to_lds_b128 %0, %1, off"
               :: "v"(lds_byte_off), "v"(gaddr)
               : "memory");
}
__device__ __forceinline__ void wait_asynccnt0() {
  asm volatile("s_wait_asynccnt 0x0" ::: "memory");
}
__device__ __forceinline__ uint32_t lds_off(const void* p) {
  return (uint32_t)(uintptr_t)p;   // low 32 bits of flat addr == LDS byte addr
}

// ---- block reductions over 256 threads ----
__device__ __forceinline__ float blk_sum(float v, float* buf) {
  const int t = threadIdx.x;
  buf[t] = v; __syncthreads();
#pragma unroll
  for (int s = 128; s > 0; s >>= 1) {
    if (t < s) buf[t] += buf[t + s];
    __syncthreads();
  }
  const float r = buf[0]; __syncthreads();
  return r;
}
__device__ __forceinline__ float blk_max(float v, float* buf) {
  const int t = threadIdx.x;
  buf[t] = v; __syncthreads();
#pragma unroll
  for (int s = 128; s > 0; s >>= 1) {
    if (t < s) buf[t] = fmaxf(buf[t], buf[t + s]);
    __syncthreads();
  }
  const float r = buf[0]; __syncthreads();
  return r;
}

// ---------------------------------------------------------------------------
// Tiled transpose + convert to bf16: out[n*ldout + k] = bf16(in[k*ldin + n]).
// ---------------------------------------------------------------------------
template <typename TIN>
__global__ __launch_bounds__(256) void transpose_to_bf16_kernel(
    const TIN* __restrict__ in, int ldin, uint16_t* __restrict__ out,
    int ldout) {
  __shared__ float tile[32][33];
  const int n0 = blockIdx.x << 5;
  const int k0 = blockIdx.y << 5;
  const int tx = threadIdx.x, ty = threadIdx.y;
#pragma unroll
  for (int i = 0; i < 4; ++i) {
    const int k = k0 + ty + (i << 3);
    tile[ty + (i << 3)][tx] = to_f32(in[(size_t)k * ldin + n0 + tx]);
  }
  __syncthreads();
#pragma unroll
  for (int i = 0; i < 4; ++i) {
    const int n = n0 + ty + (i << 3);
    out[(size_t)n * ldout + k0 + tx] = f2bf(tile[tx][ty + (i << 3)]);
  }
}

// ---------------------------------------------------------------------------
// LayerNorm over D=1024, one block (256 thr) per row, bf16 output
// ---------------------------------------------------------------------------
__global__ __launch_bounds__(256) void layernorm_bf16_kernel(
    const float* __restrict__ x, const float* __restrict__ g,
    const float* __restrict__ b, uint16_t* __restrict__ y) {
  __shared__ float buf[256];
  const int row = blockIdx.x;
  const float* xr = x + (size_t)row * DMODEL;
  float v[4]; float s = 0.f;
#pragma unroll
  for (int i = 0; i < 4; ++i) { v[i] = xr[i * 256 + threadIdx.x]; s += v[i]; }
  const float mu = blk_sum(s, buf) * (1.0f / DMODEL);
  float sq = 0.f;
#pragma unroll
  for (int i = 0; i < 4; ++i) { float d = v[i] - mu; sq += d * d; }
  const float var = blk_sum(sq, buf) * (1.0f / DMODEL);
  const float inv = rsqrtf(var + 1e-5f);
  uint16_t* yr = y + (size_t)row * DMODEL;
#pragma unroll
  for (int i = 0; i < 4; ++i) {
    const int c = i * 256 + threadIdx.x;
    yr[c] = f2bf((v[i] - mu) * inv * g[c] + b[c]);
  }
}

// ---------------------------------------------------------------------------
// Row softmax over 4096 f32 scores, P written bf16 IN PLACE (first half of row)
// ---------------------------------------------------------------------------
__global__ __launch_bounds__(256) void softmax4096_inplace_bf16(
    float* __restrict__ scores) {
  __shared__ float buf[256];
  const int row = blockIdx.x;
  float* sr = scores + (size_t)row * 4096;
  float v[16]; float m = -3.4e38f;
#pragma unroll
  for (int i = 0; i < 16; ++i) { v[i] = sr[i * 256 + threadIdx.x]; m = fmaxf(m, v[i]); }
  m = blk_max(m, buf);
  float s = 0.f;
#pragma unroll
  for (int i = 0; i < 16; ++i) { v[i] = expf(v[i] - m); s += v[i]; }
  const float inv = 1.0f / blk_sum(s, buf);
  uint16_t* pr = (uint16_t*)sr;   // bf16 P reuses the f32 row's storage
#pragma unroll
  for (int i = 0; i < 16; ++i) pr[i * 256 + threadIdx.x] = f2bf(v[i] * inv);
}

// ---------------------------------------------------------------------------
// out = enc1 + softmax(h2) over D=1024, f32
// ---------------------------------------------------------------------------
__global__ __launch_bounds__(256) void final_softmax_add(
    const float* __restrict__ h2, const float* __restrict__ enc1,
    float* __restrict__ out) {
  __shared__ float buf[256];
  const int row = blockIdx.x;
  const float* hr = h2 + (size_t)row * DMODEL;
  const float* er = enc1 + (size_t)row * DMODEL;
  float* orow = out + (size_t)row * DMODEL;
  float v[4]; float m = -3.4e38f;
#pragma unroll
  for (int i = 0; i < 4; ++i) { v[i] = hr[i * 256 + threadIdx.x]; m = fmaxf(m, v[i]); }
  m = blk_max(m, buf);
  float s = 0.f;
#pragma unroll
  for (int i = 0; i < 4; ++i) { v[i] = expf(v[i] - m); s += v[i]; }
  const float inv = 1.0f / blk_sum(s, buf);
#pragma unroll
  for (int i = 0; i < 4; ++i) {
    const int c = i * 256 + threadIdx.x;
    orow[c] = er[c] + v[i] * inv;
  }
}

// ---------------------------------------------------------------------------
// Tiled bf16 GEMM: C[M,N] = act(scale * A @ Bt^T + bias) (+ res), WMMA wave32.
//   A: [M][K] row-major bf16.  Bt: [N][K] row-major bf16 (pre-transposed).
//   block = 256 threads = 8 waves; block tile 64(M) x 128(N); K step 32.
//   waves: wm = (w&3)*16, wn = (w>>2)*64; each wave 16x64 = 4 WMMA accums.
//   LDS ping-pong, filled by global_load_async_to_lds_b128 (ASYNCcnt).
// ---------------------------------------------------------------------------
template <bool OUT_BF16, bool HAS_BIAS, bool HAS_RES, bool GELU>
__global__ __launch_bounds__(256) void gemm_wmma_bf16(
    const uint16_t* __restrict__ A, int lda,
    const uint16_t* __restrict__ Bt, int ldb,
    const float* __restrict__ bias,
    const float* __restrict__ res, int ldr,
    void* __restrict__ C, int ldc, int K, float scale) {
  __shared__ uint32_t sA[2][64 * 16];    // 64 rows x 16 k-pairs  (4 KB each)
  __shared__ uint32_t sB[2][128 * 16];   // 128 cols x 16 k-pairs (8 KB each)

  const int tid  = threadIdx.x;
  const int wave = tid >> 5;
  const int lane = tid & 31;
  const int g    = lane >> 4;        // lane group (0/1) per WMMA layout
  const int l16  = lane & 15;
  const int wm   = (wave & 3) << 4;
  const int wn   = (wave >> 2) << 6;
  const size_t m0 = (size_t)blockIdx.y << 6;
  const size_t n0 = (size_t)blockIdx.x << 7;

  const v8f vzero = {0.f, 0.f, 0.f, 0.f, 0.f, 0.f, 0.f, 0.f};
  v8f acc[4];
#pragma unroll
  for (int i = 0; i < 4; ++i) acc[i] = vzero;

  // staging assignments (each thread owns one A 16B chunk + two B 16B chunks)
  const int am = tid >> 2, ac = (tid & 3) << 3;   // A: row 0..63, col {0,8,16,24}
  const int bn = tid >> 1, bc = (tid & 1) << 4;   // B: row 0..127, col {0,16}
  const uint16_t* aptr = A + (m0 + am) * (size_t)lda + ac;
  const uint16_t* bptr = Bt + (n0 + bn) * (size_t)ldb + bc;
  const uint32_t sao = lds_off(&sA[0][am * 16 + (ac >> 1)]);
  const uint32_t sbo = lds_off(&sB[0][bn * 16 + (bc >> 1)]);

  auto stage = [&](int buf, int koff) {
    async_copy_b128(sao + buf * (64 * 16 * 4), aptr + koff);
    async_copy_b128(sbo + buf * (128 * 16 * 4), bptr + koff);
    async_copy_b128(sbo + buf * (128 * 16 * 4) + 16, bptr + koff + 8);
  };

  const int nk = K >> 5;
  stage(0, 0);
  for (int kt = 0; kt < nk; ++kt) {
    const int buf = kt & 1;
    wait_asynccnt0();            // our async LDS writes done before barrier
    __syncthreads();
    if (kt + 1 < nk) stage(buf ^ 1, (kt + 1) << 5);
    if (kt + 2 < nk)             // global_prefetch_b8 of the tile after next
      __builtin_prefetch(aptr + ((kt + 2) << 5), 0, 1);

    // A fragment 16x32 bf16 (lanes 0-15: K[0-7|16-23], lanes 16-31: K[8-15|24-31])
    const uint32_t* pa = &sA[buf][(wm + l16) * 16 + (g << 2)];
    union { uint32_t u[8]; v16bf v; } af;
#pragma unroll
    for (int v = 0; v < 4; ++v) { af.u[v] = pa[v]; af.u[v + 4] = pa[v + 8]; }

    // All four B fragments, then four back-to-back WMMAs.
    union { uint32_t u[8]; v16bf v; } bfrag[4];
#pragma unroll
    for (int nt = 0; nt < 4; ++nt) {
      const uint32_t* pb = &sB[buf][(wn + (nt << 4) + l16) * 16 + (g << 2)];
#pragma unroll
      for (int j = 0; j < 8; ++j) bfrag[nt].u[j] = pb[j < 4 ? j : j + 4];
    }
#pragma unroll
    for (int nt = 0; nt < 4; ++nt)
      acc[nt] = __builtin_amdgcn_wmma_f32_16x16x32_bf16(
          false, af.v, false, bfrag[nt].v, (short)0, acc[nt], false, false);
  }

  // Epilogue. C/D layout: VGPR v -> row 8g+v, lane%16 -> col.
#pragma unroll
  for (int nt = 0; nt < 4; ++nt) {
    const size_t col = n0 + wn + (nt << 4) + l16;
    const float bv = HAS_BIAS ? bias[col] : 0.0f;
#pragma unroll
    for (int v = 0; v < 8; ++v) {
      const size_t row = m0 + wm + (g << 3) + v;
      float val = acc[nt][v] * scale + bv;
      if (GELU) val = gelu_exact(val);
      if (HAS_RES) val += res[row * (size_t)ldr + col];
      if (OUT_BF16)
        ((uint16_t*)C)[row * (size_t)ldc + col] = f2bf(val);
      else
        ((float*)C)[row * (size_t)ldc + col] = val;
    }
  }
}

// ---------------------------------------------------------------------------
// Workspace layout (bytes, all offsets 256B-aligned). Total = 165,675,008 B.
// ---------------------------------------------------------------------------
static constexpr size_t OFF_WQ   = 0;                          // wqT  [3072][1024] bf16
static constexpr size_t OFF_WM   = OFF_WQ   + 6291456;         // wmT  [1024][1024] bf16
static constexpr size_t OFF_WH1  = OFF_WM   + 2097152;         // wh1T [512][1024] bf16
static constexpr size_t OFF_WH2  = OFF_WH1  + 1048576;         // wh2T [1024][512] bf16
static constexpr size_t OFF_LN   = OFF_WH2  + 1048576;         // 4096*1024 bf16 (ln1 then ln2)
static constexpr size_t OFF_QKV  = OFF_LN   + 8388608;         // 4096*3072 bf16
static constexpr size_t OFF_SC   = OFF_QKV  + 25165824;        // 4096*4096 f32 (P bf16 in place)
static constexpr size_t OFF_AV   = OFF_SC   + 67108864;        // 4096*1024 bf16
static constexpr size_t OFF_ENC1 = OFF_AV   + 8388608;         // 4096*1024 f32
static constexpr size_t OFF_H1   = OFF_ENC1 + 16777216;        // 4096*512 bf16
static constexpr size_t OFF_H2   = OFF_H1   + 4194304;         // 4096*1024 f32
static constexpr size_t OFF_VT   = OFF_H2   + 16777216;        // vT [1024][4096] bf16

extern "C" void kernel_launch(void* const* d_in, const int* in_sizes, int n_in,
                              void* d_out, int out_size, void* d_ws,
                              size_t ws_size, hipStream_t stream) {
  const float* x     = (const float*)d_in[0];
  const float* w_qkv = (const float*)d_in[1];
  const float* b_qkv = (const float*)d_in[2];
  const float* w_msa = (const float*)d_in[3];
  const float* b_msa = (const float*)d_in[4];
  const float* ln_g  = (const float*)d_in[5];
  const float* ln_b  = (const float*)d_in[6];
  const float* w_h1  = (const float*)d_in[7];
  const float* b_h1  = (const float*)d_in[8];
  const float* w_h2  = (const float*)d_in[9];
  const float* b_h2  = (const float*)d_in[10];
  float* out = (float*)d_out;

  char* ws = (char*)d_ws;
  uint16_t* wqT   = (uint16_t*)(ws + OFF_WQ);
  uint16_t* wmT   = (uint16_t*)(ws + OFF_WM);
  uint16_t* wh1T  = (uint16_t*)(ws + OFF_WH1);
  uint16_t* wh2T  = (uint16_t*)(ws + OFF_WH2);
  uint16_t* ln_bf = (uint16_t*)(ws + OFF_LN);
  uint16_t* qkv_bf= (uint16_t*)(ws + OFF_QKV);
  float*    scores= (float*)   (ws + OFF_SC);
  uint16_t* av_bf = (uint16_t*)(ws + OFF_AV);
  float*    enc1  = (float*)   (ws + OFF_ENC1);
  uint16_t* h1_bf = (uint16_t*)(ws + OFF_H1);
  float*    h2    = (float*)   (ws + OFF_H2);
  uint16_t* vT    = (uint16_t*)(ws + OFF_VT);

  const dim3 tb(32, 8);

  // 1) weights f32 -> bf16, transposed to [N][K]
  transpose_to_bf16_kernel<float><<<dim3(QKV_N / 32, DMODEL / 32), tb, 0, stream>>>(
      w_qkv, QKV_N, wqT, DMODEL);
  transpose_to_bf16_kernel<float><<<dim3(DMODEL / 32, DMODEL / 32), tb, 0, stream>>>(
      w_msa, DMODEL, wmT, DMODEL);
  transpose_to_bf16_kernel<float><<<dim3(HID_ / 32, DMODEL / 32), tb, 0, stream>>>(
      w_h1, HID_, wh1T, DMODEL);
  transpose_to_bf16_kernel<float><<<dim3(DMODEL / 32, HID_ / 32), tb, 0, stream>>>(
      w_h2, DMODEL, wh2T, HID_);

  // 2) ln1 = LN(x) -> bf16
  layernorm_bf16_kernel<<<N_TOK, 256, 0, stream>>>(x, ln_g, ln_b, ln_bf);

  // 3) qkv = ln1 @ w_qkv + b_qkv  (4096x3072, K=1024) -> bf16
  gemm_wmma_bf16<true, true, false, false>
      <<<dim3(QKV_N / 128, N_TOK / 64), 256, 0, stream>>>(
          ln_bf, DMODEL, wqT, DMODEL, b_qkv, nullptr, 0, qkv_bf, QKV_N,
          DMODEL, 1.0f);

  // 3b) vT = transpose(v)  ([4096][1024] bf16 -> [1024][4096] bf16)
  transpose_to_bf16_kernel<uint16_t><<<dim3(DMODEL / 32, N_TOK / 32), tb, 0, stream>>>(
      qkv_bf + 2 * MHK_, QKV_N, vT, N_TOK);

  // 4) scores = (q @ k^T) / 8  (4096x4096, K=1024); k rows already [N][K]
  gemm_wmma_bf16<false, false, false, false>
      <<<dim3(N_TOK / 128, N_TOK / 64), 256, 0, stream>>>(
          qkv_bf, QKV_N, qkv_bf + MHK_, QKV_N, nullptr, nullptr, 0, scores,
          N_TOK, DMODEL, 0.125f);

  // 5) P = softmax(scores) rows -> bf16 in place (lda = 8192 bf16)
  softmax4096_inplace_bf16<<<N_TOK, 256, 0, stream>>>(scores);

  // 6) attnV = P @ v  (4096x1024, K=4096) via vT
  gemm_wmma_bf16<true, false, false, false>
      <<<dim3(DMODEL / 128, N_TOK / 64), 256, 0, stream>>>(
          (const uint16_t*)scores, 2 * N_TOK, vT, N_TOK, nullptr, nullptr, 0,
          av_bf, DMODEL, N_TOK, 1.0f);

  // 7) enc1 = x + attnV @ w_msa + b_msa  (4096x1024, K=1024) -> f32
  gemm_wmma_bf16<false, true, true, false>
      <<<dim3(DMODEL / 128, N_TOK / 64), 256, 0, stream>>>(
          av_bf, DMODEL, wmT, DMODEL, b_msa, x, DMODEL, enc1, DMODEL, DMODEL,
          1.0f);

  // 8) ln2 = LN(enc1) -> bf16 (reuse ln buffer)
  layernorm_bf16_kernel<<<N_TOK, 256, 0, stream>>>(enc1, ln_g, ln_b, ln_bf);

  // 9) h1 = gelu(ln2 @ w_h1 + b_h1)  (4096x512, K=1024) -> bf16
  gemm_wmma_bf16<true, true, false, true>
      <<<dim3(HID_ / 128, N_TOK / 64), 256, 0, stream>>>(
          ln_bf, DMODEL, wh1T, DMODEL, b_h1, nullptr, 0, h1_bf, HID_, DMODEL,
          1.0f);

  // 10) h2 = gelu(h1 @ w_h2 + b_h2)  (4096x1024, K=512) -> f32
  gemm_wmma_bf16<false, true, false, true>
      <<<dim3(DMODEL / 128, N_TOK / 64), 256, 0, stream>>>(
          h1_bf, HID_, wh2T, HID_, b_h2, nullptr, 0, h2, DMODEL, HID_, 1.0f);

  // 11) out = enc1 + softmax(h2)
  final_softmax_add<<<N_TOK, 256, 0, stream>>>(h2, enc1, out);
}
